// QLSTMHybrid_65481071398078
// MI455X (gfx1250) — matrix operand order repaired and checked by reading, not compiled, not run on hardware
//
#include <hip/hip_runtime.h>

typedef __attribute__((ext_vector_type(16))) _Float16 v16h;
typedef __attribute__((ext_vector_type(8)))  float    v8f;

#define SEQ   2048
#define BATCH 256
#define ED    16
#define HD    16
#define NW    16
#define TAGS  64

__device__ __forceinline__ v8f wmma16(v16h a, v16h b, v8f c) {
    // v_wmma_f32_16x16x32_f16  D = A(16x32 f16) * B(32x16 f16) + C(16x16 f32)
    return __builtin_amdgcn_wmma_f32_16x16x32_f16(false, a, false, b, (short)0, c, false, false);
}

__device__ __forceinline__ float fsig(float x)  { return 1.0f / (1.0f + __expf(-x)); }
__device__ __forceinline__ float ftanh(float x) { return 2.0f / (1.0f + __expf(-2.0f * x)) - 1.0f; }

union HV { float4 v; _Float16 h[8]; };

// Load 16 consecutive f32 weights into a WMMA-B fragment in table order:
// slots 0..7 of lane half hi carry K = 8*hi + e; slots 8..15 (K=16..31) are zero-padded.
// Loads are UNCONDITIONAL (address valid for every lane).
__device__ __forceinline__ v16h loadB_halfK(const float* __restrict__ p, float sel) {
    v16h B;
    (void)sel;
    const int hk = 8 * ((threadIdx.x >> 4) & 1);
    #pragma unroll
    for (int e = 0; e < 8; ++e) B[e] = (_Float16)p[hk + e];
    #pragma unroll
    for (int e = 8; e < 16; ++e) B[e] = (_Float16)0.0f;
    return B;
}

// ---- Phase 1: encoded = cos(emb[sentence]@w_eq.T + b + th); then per-gate x-projection
//      zx_g = encoded @ Wx_g.T + b_g + th_g   (bias folded into WMMA C accumulator)
//      zx layout (f16): [tile][gate][lane][8]  -> 16B contiguous per lane, fully coalesced
__global__ __launch_bounds__(256) void k_encode(
    const int* __restrict__ sentence, const float* __restrict__ emb,
    const float* __restrict__ w_eq, const float* __restrict__ b_eq, const float* __restrict__ th,
    const float* __restrict__ w_f, const float* __restrict__ b_f, const float* __restrict__ t_f,
    const float* __restrict__ w_i, const float* __restrict__ b_i, const float* __restrict__ t_i,
    const float* __restrict__ w_u, const float* __restrict__ b_u, const float* __restrict__ t_u,
    const float* __restrict__ w_o, const float* __restrict__ b_o, const float* __restrict__ t_o,
    _Float16* __restrict__ zx)
{
    __shared__ _Float16 esh[8][16][16];            // per-warp encoded tile [M][N]
    const int lane = threadIdx.x & 31;
    const int w    = threadIdx.x >> 5;
    const int tile = blockIdx.x * 8 + w;           // 16 (s,b) rows per tile
    const int r    = lane & 15;
    const int hi   = lane >> 4;
    const float sel = (hi == 0) ? 1.0f : 0.0f;

    // B = w_eq^T : lanes 0-15 carry K 0-7, lanes 16-31 K 8-15 (slots 0-7); slots 8-15 zero-pad K 16-31
    const v16h Beq = loadB_halfK(w_eq + r * ED, sel);

    const int row0 = tile * 16;
    const int idx  = sentence[row0 + r];
    const float* ep = emb + (size_t)idx * ED + 8 * hi;
    v16h A;
    #pragma unroll
    for (int e = 0; e < 8; ++e)  A[e] = (_Float16)ep[e];      // K = e + 8*hi
    #pragma unroll
    for (int e = 8; e < 16; ++e) A[e] = (_Float16)0.0f;       // K = 16..31 pad

    v8f Cb;
    const float bneq = b_eq[r] + th[r];                        // indexed by N = r
    #pragma unroll
    for (int i = 0; i < 8; ++i) Cb[i] = bneq;

    const v8f D = wmma16(A, Beq, Cb);
    #pragma unroll
    for (int i = 0; i < 8; ++i)
        esh[w][i + 8 * hi][r] = (_Float16)__cosf(D[i]);        // D-layout -> [M][N]
    __syncthreads();

    // A = encoded tile in A-layout (K = hidden unit 0..15, pad 16..31)
    v16h Ae;
    #pragma unroll
    for (int e = 0; e < 8; ++e)  Ae[e] = esh[w][r][8 * hi + e];
    #pragma unroll
    for (int e = 8; e < 16; ++e) Ae[e] = (_Float16)0.0f;

    const float* wg[4] = { w_f, w_i, w_u, w_o };
    const float* bg[4] = { b_f, b_i, b_u, b_o };
    const float* tg[4] = { t_f, t_i, t_u, t_o };
    #pragma unroll
    for (int g = 0; g < 4; ++g) {
        const v16h Bx = loadB_halfK(wg[g] + r * 32, sel);      // x columns 0..15 of w_g

        v8f Cg;
        const float bn = bg[g][r] + tg[g][r];
        #pragma unroll
        for (int i = 0; i < 8; ++i) Cg[i] = bn;

        const v8f Z = wmma16(Ae, Bx, Cg);
        HV u;
        #pragma unroll
        for (int i = 0; i < 8; ++i) u.h[i] = (_Float16)Z[i];
        ((float4*)zx)[((size_t)tile * 4 + g) * 32 + lane] = u.v;
    }
}

// ---- Phase 2: sequential scan. Recurrent work per step = LDS h-transpose + 4 WMMAs
//      with C = prefetched zx (double-buffered), + transcendentals. One wave / batch tile.
__global__ __launch_bounds__(32) void k_lstm(
    const _Float16* __restrict__ zx,
    const float* __restrict__ w_f, const float* __restrict__ w_i,
    const float* __restrict__ w_u, const float* __restrict__ w_o,
    _Float16* __restrict__ hout)
{
    __shared__ _Float16 hsh[16][16];               // h tile [M][N], f16
    const int lane = threadIdx.x;
    const int r    = lane & 15;
    const int hi   = lane >> 4;
    const int bt   = blockIdx.x;                   // batch tile
    const float sel = (hi == 0) ? 1.0f : 0.0f;

    // Recurrent weights Wh_g = w_g[:,16:32] in B layout (K 0..15 real, 16..31 pad)
    v16h Bh[4];
    const float* wg[4] = { w_f, w_i, w_u, w_o };
    #pragma unroll
    for (int g = 0; g < 4; ++g)
        Bh[g] = loadB_halfK(wg[g] + r * 32 + 16, sel);

    v16h A;                                        // slots 8..15 permanently zero
    #pragma unroll
    for (int e = 0; e < 16; ++e) A[e] = (_Float16)0.0f;

    float cc[8];
    #pragma unroll
    for (int i = 0; i < 8; ++i) cc[i] = 0.0f;
    #pragma unroll
    for (int i = 0; i < 8; ++i) hsh[i + 8 * hi][r] = (_Float16)0.0f;
    __syncthreads();

    const float4* zx4 = (const float4*)zx;
    float4 cur[4], nxt[4];
    #pragma unroll
    for (int g = 0; g < 4; ++g)
        cur[g] = zx4[((size_t)bt * 4 + g) * 32 + lane];        // s=0 : T = bt

    for (int s = 0; s < SEQ; ++s) {
        // prefetch next step's pre-activations (independent of recurrence)
        const int sn = (s + 1 < SEQ) ? s + 1 : s;
        const size_t Tn = (size_t)sn * 16 + bt;
        #pragma unroll
        for (int g = 0; g < 4; ++g)
            nxt[g] = zx4[(Tn * 4 + g) * 32 + lane];

        // A low half = previous h from LDS (row r, cols 8*hi .. 8*hi+7)
        #pragma unroll
        for (int e = 0; e < 8; ++e) A[e] = hsh[r][8 * hi + e];

        v8f Z[4];
        #pragma unroll
        for (int g = 0; g < 4; ++g) {
            HV u; u.v = cur[g];
            v8f C;
            #pragma unroll
            for (int i = 0; i < 8; ++i) C[i] = (float)u.h[i];
            Z[g] = wmma16(A, Bh[g], C);            // z_g = h@Wh^T + (x@Wx^T + b + th)
        }

        __syncthreads();                            // h reads done before overwrite
        _Float16 hh[8];
        #pragma unroll
        for (int i = 0; i < 8; ++i) {
            const float f  = fsig (__cosf(Z[0][i]));
            const float ig = fsig (__cosf(Z[1][i]));
            const float gg = ftanh(__cosf(Z[2][i]));
            const float o  = fsig (__cosf(Z[3][i]));
            const float c  = f * cc[i] + ig * gg;
            cc[i] = c;
            hh[i] = (_Float16)(o * ftanh(c));
        }
        #pragma unroll
        for (int i = 0; i < 8; ++i) hsh[i + 8 * hi][r] = hh[i];
        __syncthreads();

        const size_t rowb = (size_t)s * BATCH + bt * 16;       // row-major (S,B,HD) f16
        #pragma unroll
        for (int i = 0; i < 8; ++i)
            hout[(rowb + i + 8 * hi) * HD + r] = hh[i];

        #pragma unroll
        for (int g = 0; g < 4; ++g) cur[g] = nxt[g];
    }
}

// ---- Phase 3: logits = h @ w_tag.T + b_tag ; log_softmax over TAGS=64 ----------------
__global__ __launch_bounds__(256) void k_tag(
    const _Float16* __restrict__ hin, const float* __restrict__ w_tag,
    const float* __restrict__ b_tag, float* __restrict__ out)
{
    const int lane = threadIdx.x & 31;
    const int tile = blockIdx.x * 8 + (threadIdx.x >> 5);
    const int r    = lane & 15;
    const int hi   = lane >> 4;
    const float sel = (hi == 0) ? 1.0f : 0.0f;

    v16h Bt[4];
    #pragma unroll
    for (int tt = 0; tt < 4; ++tt)
        Bt[tt] = loadB_halfK(w_tag + (tt * 16 + r) * HD, sel);

    const int row0 = tile * 16;
    HV u;                                                      // 16B contiguous f16 load
    u.v = ((const float4*)hin)[(size_t)(row0 + r) * 2 + hi];
    v16h A;
    #pragma unroll
    for (int e = 0; e < 8; ++e)  A[e] = u.h[e];
    #pragma unroll
    for (int e = 8; e < 16; ++e) A[e] = (_Float16)0.0f;

    const v8f z0 = {};
    v8f zt[4];
    #pragma unroll
    for (int tt = 0; tt < 4; ++tt) zt[tt] = wmma16(A, Bt[tt], z0);

    float bb[4];
    #pragma unroll
    for (int tt = 0; tt < 4; ++tt) bb[tt] = b_tag[tt * 16 + r];

    #pragma unroll
    for (int i = 0; i < 8; ++i) {
        float v[4];
        #pragma unroll
        for (int tt = 0; tt < 4; ++tt) v[tt] = zt[tt][i] + bb[tt];

        float m = fmaxf(fmaxf(v[0], v[1]), fmaxf(v[2], v[3]));
        #pragma unroll
        for (int off = 1; off < 16; off <<= 1)
            m = fmaxf(m, __shfl_xor(m, off, 32));              // within 16-lane half

        float sum = __expf(v[0] - m) + __expf(v[1] - m) + __expf(v[2] - m) + __expf(v[3] - m);
        #pragma unroll
        for (int off = 1; off < 16; off <<= 1)
            sum += __shfl_xor(sum, off, 32);

        const float lse = m + __logf(sum);
        const size_t row = (size_t)(row0 + i + 8 * hi);
        #pragma unroll
        for (int tt = 0; tt < 4; ++tt)
            out[row * TAGS + tt * 16 + r] = v[tt] - lse;
    }
}

extern "C" void kernel_launch(void* const* d_in, const int* in_sizes, int n_in,
                              void* d_out, int out_size, void* d_ws, size_t ws_size,
                              hipStream_t stream) {
    (void)in_sizes; (void)n_in; (void)out_size; (void)ws_size;
    const int*   sentence = (const int*)  d_in[0];
    const float* emb      = (const float*)d_in[1];
    const float* w_eq     = (const float*)d_in[2];
    const float* b_eq     = (const float*)d_in[3];
    const float* th_conv  = (const float*)d_in[4];
    const float* w_f = (const float*)d_in[5];  const float* b_f = (const float*)d_in[6];  const float* t_f = (const float*)d_in[7];
    const float* w_i = (const float*)d_in[8];  const float* b_i = (const float*)d_in[9];  const float* t_i = (const float*)d_in[10];
    const float* w_u = (const float*)d_in[11]; const float* b_u = (const float*)d_in[12]; const float* t_u = (const float*)d_in[13];
    const float* w_o = (const float*)d_in[14]; const float* b_o = (const float*)d_in[15]; const float* t_o = (const float*)d_in[16];
    const float* w_tag = (const float*)d_in[17];
    const float* b_tag = (const float*)d_in[18];

    _Float16* zxb  = (_Float16*)d_ws;                                        // 64 MB
    _Float16* hbuf = (_Float16*)((char*)d_ws +
                     (size_t)SEQ * BATCH * 4 * 16 * sizeof(_Float16));       // 16 MB
    float* out = (float*)d_out;

    const int tiles = (SEQ * BATCH) / 16;   // 32768 row-tiles

    k_encode<<<tiles / 8, 256, 0, stream>>>(sentence, emb, w_eq, b_eq, th_conv,
                                            w_f, b_f, t_f, w_i, b_i, t_i,
                                            w_u, b_u, t_u, w_o, b_o, t_o, zxb);
    k_lstm<<<BATCH / 16, 32, 0, stream>>>(zxb, w_f, w_i, w_u, w_o, hbuf);
    k_tag<<<tiles / 8, 256, 0, stream>>>(hbuf, w_tag, b_tag, out);
}
